// LlamaDecoderLayer_53824530153490
// MI455X (gfx1250) — matrix operand order repaired and checked by reading, not compiled, not run on hardware
//
// Mixtral-style decoder layer for MI455X (gfx1250, wave32, WMMA + TDM).
// All matmuls route through v_wmma_f32_16x16x32_bf16 (fp32 accumulate).
// GEMM tiles are staged into LDS by the Tensor Data Mover (tensor_load_to_lds,
// double-buffered, TENSORcnt-pipelined) so the VMEM/DS pipes stay free for
// fragment reads feeding the WMMA stream.
#include <hip/hip_runtime.h>

typedef __attribute__((ext_vector_type(16))) __bf16        v16bf;
typedef __attribute__((ext_vector_type(8)))  float         v8f;
typedef __attribute__((ext_vector_type(4)))  unsigned int  u32x4;
typedef __attribute__((ext_vector_type(8)))  int           i32x8;
typedef __attribute__((ext_vector_type(4)))  int           i32x4;

union Frag { v16bf v; uint4 q[2]; };

__device__ __forceinline__ v8f wmma_bf16(v16bf a, v16bf b, v8f c) {
  // (neg_a, A, neg_b, B, c_mod, C, reuse_a, reuse_b)
  return __builtin_amdgcn_wmma_f32_16x16x32_bf16(false, a, false, b, (short)0, c,
                                                 false, false);
}

#if defined(__AMDGCN__) && __has_builtin(__builtin_amdgcn_tensor_load_to_lds)
#define USE_TDM 1
#else
#define USE_TDM 0
#endif

constexpr int kB  = 2;
constexpr int kS  = 2048;
constexpr int kD  = 2048;
constexpr int kHQ = 16;
constexpr int kHKV = 8;
constexpr int kHD = 128;
constexpr int kE  = 8;
constexpr int kF  = 4096;
constexpr int kT  = kB * kS;                      // 4096 tokens
constexpr int kQKV = (kHQ + 2 * kHKV) * kHD;      // 4096 fused qkv cols
constexpr float kEps = 1e-6f;

#if USE_TDM
// ---------------------------------------------------------------------------
// Issue one TDM 2-D tile load: bf16 tile [tile_rows x tile_k] from a row-major
// [rows x K] tensor (row stride = K elements) into LDS at lds_off (bytes).
// D# layout per CDNA5 ISA:
//   group0: count=1 | lds_addr | global_addr[56:0] | type=2
//   group1: data_size=1(2B) | tensor_dim0=K | tensor_dim1=rows |
//           tile_dim0, tile_dim1 | tensor_dim0_stride=K
// Groups 2/3 + trailing vector operand are zero (2-D tile, no gather/iterate).
// Executed by one wave only (EXEC is ignored by TDM); tracked with TENSORcnt.
// This toolchain's builtin takes 6 args:
//   (uint32x4 g0, int32x8 g1, int32x4 g2, int32x4 g3, int32x8, i32 cpol)
// ---------------------------------------------------------------------------
__device__ __forceinline__ void tdm_load_tile_bf16(unsigned int lds_off,
                                                   const void* gaddr,
                                                   unsigned int K_elems,
                                                   unsigned int rows,
                                                   unsigned int tile_k,
                                                   unsigned int tile_rows) {
  const unsigned long long ga = (unsigned long long)gaddr;
  u32x4 g0;
  g0[0] = 1u;                                         // count=1, user descriptor
  g0[1] = lds_off;                                    // LDS byte address
  g0[2] = (unsigned int)(ga & 0xffffffffull);         // global_addr[31:0]
  g0[3] = (unsigned int)((ga >> 32) & 0x01ffffffull)  // global_addr[56:32]
          | (2u << 30);                               // type = 2 ("image")
  i32x8 g1;
  g1[0] = (int)(1u << 16);                            // data_size = 1 -> 2 bytes
  g1[1] = (int)((K_elems & 0xffffu) << 16);           // tensor_dim0[15:0]
  g1[2] = (int)((K_elems >> 16) | ((rows & 0xffffu) << 16));
  g1[3] = (int)((rows >> 16) | (tile_k << 16));       // tile_dim0
  g1[4] = (int)tile_rows;                             // tile_dim1 (tile_dim2=0)
  g1[5] = (int)K_elems;                               // tensor_dim0_stride lo
  g1[6] = 0;                                          // stride hi / dim1_stride
  g1[7] = 0;
  const i32x4 z4 = {0, 0, 0, 0};                      // groups 2/3 unused (2-D)
  const i32x8 z8 = {0, 0, 0, 0, 0, 0, 0, 0};
  __builtin_amdgcn_tensor_load_to_lds(g0, g1, z4, z4, z8, 0);
}
#endif

// ---------------------------------------------------------------------------
// RMSNorm: fp32 in -> bf16 out (one block per token row)
// ---------------------------------------------------------------------------
__global__ __launch_bounds__(256)
void rmsnorm_to_bf16_kernel(const float* __restrict__ x, const float* __restrict__ g,
                            __bf16* __restrict__ out, int D) {
  const int row = blockIdx.x;
  const float* xr = x + (size_t)row * D;
  float ss = 0.f;
  for (int d = threadIdx.x; d < D; d += 256) { float v = xr[d]; ss += v * v; }
  for (int m = 16; m >= 1; m >>= 1) ss += __shfl_xor(ss, m, 32);
  __shared__ float red[8];
  const int wave = threadIdx.x >> 5, lane = threadIdx.x & 31;
  if (lane == 0) red[wave] = ss;
  __syncthreads();
  if (wave == 0) {
    float v = (lane < 8) ? red[lane] : 0.f;
    for (int m = 4; m >= 1; m >>= 1) v += __shfl_xor(v, m, 32);
    if (lane == 0) red[0] = v;
  }
  __syncthreads();
  const float inv = rsqrtf(red[0] / (float)D + kEps);
  for (int d = threadIdx.x; d < D; d += 256)
    out[(size_t)row * D + d] = (__bf16)(xr[d] * inv * g[d]);
}

// ---------------------------------------------------------------------------
// fp32 [K,N] row-major  ->  bf16 W^T [N,K] row-major  (tiled, LDS transpose)
// ---------------------------------------------------------------------------
__global__ __launch_bounds__(256)
void transpose_f32_to_bf16_kernel(const float* __restrict__ W, __bf16* __restrict__ Wt,
                                  int K, int N) {
  __shared__ float tile[32][33];
  const int tx = threadIdx.x, ty = threadIdx.y;     // 32 x 8
  const int k0 = blockIdx.y * 32, n0 = blockIdx.x * 32;
  #pragma unroll
  for (int i = 0; i < 32; i += 8)
    tile[ty + i][tx] = W[(size_t)(k0 + ty + i) * N + n0 + tx];
  __syncthreads();
  #pragma unroll
  for (int i = 0; i < 32; i += 8)
    Wt[(size_t)(n0 + ty + i) * K + k0 + tx] = (__bf16)tile[tx][ty + i];
}

// ---------------------------------------------------------------------------
// Generic WMMA GEMM:  C[M,N](fp32) = A[M,K](bf16, row-major) * Bt[N,K]^T
// Block = 256 thr (8 waves), tile 128x128, BK=32.  Wave grid 4(M) x 2(N);
// each wave owns a 32x64 sub-tile = 2x4 accum tiles, 8 v_wmma per K-step.
// TDM path: wave 0 DMAs the next K-slab (A+B tiles) into the ping-pong LDS
// buffer while all waves compute from the current one; stage retirement via
// s_wait_tensorcnt (TENSORcnt completes in-order per wave).
// M,N multiples of 128; K multiple of 32 (true for all call sites here).
// ---------------------------------------------------------------------------
__global__ __launch_bounds__(256)
void gemm_bf16_wmma_kernel(const __bf16* __restrict__ A, const __bf16* __restrict__ Bt,
                           float* __restrict__ C, int M, int N, int K) {
  __shared__ __bf16 sA[2][128][32];
  __shared__ __bf16 sB[2][128][32];
  const int tid  = threadIdx.x;
  const int lane = tid & 31;
  const int wave = tid >> 5;
  const int wm   = (wave & 3) * 32;
  const int wn   = (wave >> 2) * 64;
  const int m0   = blockIdx.y * 128;
  const int n0   = blockIdx.x * 128;
  const int mloc = lane & 15;
  const int kh   = lane >> 4;

  v8f zf = {};
  v8f acc[2][4];
  #pragma unroll
  for (int i = 0; i < 2; ++i)
    #pragma unroll
    for (int j = 0; j < 4; ++j) acc[i][j] = zf;

  const int nk = K >> 5;

#if USE_TDM
  const unsigned int ldsA0 = (unsigned int)(unsigned long long)&sA[0][0][0];
  const unsigned int ldsA1 = (unsigned int)(unsigned long long)&sA[1][0][0];
  const unsigned int ldsB0 = (unsigned int)(unsigned long long)&sB[0][0][0];
  const unsigned int ldsB1 = (unsigned int)(unsigned long long)&sB[1][0][0];
  if (tid < 32) {   // wave 0 is the TDM issuer (EXEC ignored by TDM)
    tdm_load_tile_bf16(ldsA0, A  + (size_t)m0 * K, (unsigned)K,
                       (unsigned)(M - m0), 32u, 128u);
    tdm_load_tile_bf16(ldsB0, Bt + (size_t)n0 * K, (unsigned)K,
                       (unsigned)(N - n0), 32u, 128u);
  }
#else
  const int ldRow = tid >> 1;          // 0..127
  const int ldSeg = (tid & 1) * 16;    // 0 or 16 (elements)
#endif

  for (int ki = 0; ki < nk; ++ki) {
    const int buf = ki & 1;
#if USE_TDM
    if (tid < 32) {
      if (ki + 1 < nk) {
        const int kn = (ki + 1) << 5;
        tdm_load_tile_bf16(buf ? ldsA0 : ldsA1, A + (size_t)m0 * K + kn,
                           (unsigned)K, (unsigned)(M - m0), 32u, 128u);
        tdm_load_tile_bf16(buf ? ldsB0 : ldsB1, Bt + (size_t)n0 * K + kn,
                           (unsigned)K, (unsigned)(N - n0), 32u, 128u);
        __builtin_amdgcn_s_wait_tensorcnt(2);   // retire current stage's pair
      } else {
        __builtin_amdgcn_s_wait_tensorcnt(0);
      }
    }
    __syncthreads();                            // LDS stage visible to all waves
#else
    __syncthreads();
    {
      const int k0 = ki << 5;
      const __bf16* ga = A  + (size_t)(m0 + ldRow) * K + k0 + ldSeg;
      const __bf16* gb = Bt + (size_t)(n0 + ldRow) * K + k0 + ldSeg;
      *(uint4*)&sA[buf][ldRow][ldSeg]     = *(const uint4*)ga;
      *(uint4*)&sA[buf][ldRow][ldSeg + 8] = *(const uint4*)(ga + 8);
      *(uint4*)&sB[buf][ldRow][ldSeg]     = *(const uint4*)gb;
      *(uint4*)&sB[buf][ldRow][ldSeg + 8] = *(const uint4*)(gb + 8);
    }
    __syncthreads();
#endif

    Frag a[2], b[4];
    #pragma unroll
    for (int i = 0; i < 2; ++i) {      // A frag: M=lane%16, K segs per ISA layout
      a[i].q[0] = *(const uint4*)&sA[buf][wm + i * 16 + mloc][kh * 8];
      a[i].q[1] = *(const uint4*)&sA[buf][wm + i * 16 + mloc][16 + kh * 8];
    }
    #pragma unroll
    for (int j = 0; j < 4; ++j) {      // B frag: N=lane%16, K segs per ISA layout
      b[j].q[0] = *(const uint4*)&sB[buf][wn + j * 16 + mloc][kh * 8];
      b[j].q[1] = *(const uint4*)&sB[buf][wn + j * 16 + mloc][16 + kh * 8];
    }
    #pragma unroll
    for (int i = 0; i < 2; ++i)
      #pragma unroll
      for (int j = 0; j < 4; ++j)
        acc[i][j] = wmma_bf16(a[i].v, b[j].v, acc[i][j]);

#if USE_TDM
    __syncthreads();   // all reads of `buf` done before it is re-DMAed next iter
#endif
  }

  #pragma unroll
  for (int i = 0; i < 2; ++i)
    #pragma unroll
    for (int j = 0; j < 4; ++j)
      #pragma unroll
      for (int r = 0; r < 8; ++r)      // C layout: col=lane%16, rows (lane/16)*8+r
        C[(size_t)(m0 + wm + i * 16 + kh * 8 + r) * N + n0 + wn + j * 16 + mloc] =
            acc[i][j][r];
}

// ---------------------------------------------------------------------------
// RoPE + head split: qkv fp32 [T, 4096] -> q/k/v bf16 in [B,H,S,HD] layout.
// q additionally scaled by HD^-0.5 (folded here so attention scores need none).
// ---------------------------------------------------------------------------
__global__ __launch_bounds__(256)
void rope_split_kernel(const float* __restrict__ qkv, const int* __restrict__ pos,
                       __bf16* __restrict__ qb, __bf16* __restrict__ kb,
                       __bf16* __restrict__ vb) {
  const int row = blockIdx.x;                 // token index
  const int b = row / kS, s = row % kS;
  const float p = (float)pos[row];
  const float* src = qkv + (size_t)row * kQKV;
  constexpr float kLogThetaOverHalf = 0.14391156f;   // ln(10000)/64
  constexpr float kQScale = 0.08838834764831845f;    // 128^-0.5

  for (int idx = threadIdx.x; idx < kHQ * 64; idx += 256) {
    const int hh = idx >> 6, i = idx & 63;
    const float x1 = src[hh * kHD + i], x2 = src[hh * kHD + i + 64];
    float sn, cs;
    __sincosf(p * __expf(-(float)i * kLogThetaOverHalf), &sn, &cs);
    const size_t o = (((size_t)b * kHQ + hh) * kS + s) * kHD + i;
    qb[o]      = (__bf16)((x1 * cs - x2 * sn) * kQScale);
    qb[o + 64] = (__bf16)((x2 * cs + x1 * sn) * kQScale);
  }
  for (int idx = threadIdx.x; idx < kHKV * 64; idx += 256) {
    const int hh = idx >> 6, i = idx & 63;
    const float x1 = src[kD + hh * kHD + i], x2 = src[kD + hh * kHD + i + 64];
    float sn, cs;
    __sincosf(p * __expf(-(float)i * kLogThetaOverHalf), &sn, &cs);
    const size_t o = (((size_t)b * kHKV + hh) * kS + s) * kHD + i;
    kb[o]      = (__bf16)(x1 * cs - x2 * sn);
    kb[o + 64] = (__bf16)(x2 * cs + x1 * sn);
  }
  for (int idx = threadIdx.x; idx < kHKV * kHD; idx += 256) {
    const int hh = idx >> 7, dd = idx & 127;
    vb[(((size_t)b * kHKV + hh) * kS + s) * kHD + dd] =
        (__bf16)src[kD + kHKV * kHD + idx];
  }
}

// ---------------------------------------------------------------------------
// Flash attention (GQA, causal + segment mask), all-WMMA.
// Grid: (B*HQ, S/64). Block: 128 thr = 4 waves; each wave owns 16 q rows.
// KV chunks of 32: 8 WMMA for S = Q*K^T, 8 WMMA for ctx += P*V per wave.
// Online softmax in fp32 using the C-matrix lane layout directly.
// ---------------------------------------------------------------------------
__global__ __launch_bounds__(128)
void flash_attn_kernel(const __bf16* __restrict__ Q, const __bf16* __restrict__ Kc,
                       const __bf16* __restrict__ Vc, const int* __restrict__ seg,
                       __bf16* __restrict__ ctx) {
  const int bh = blockIdx.x;
  const int b = bh / kHQ, h = bh % kHQ;
  const int hk = h / (kHQ / kHKV);
  const int qt = blockIdx.y;
  const int wave = threadIdx.x >> 5;
  const int lane = threadIdx.x & 31;
  const int mloc = lane & 15, kh = lane >> 4;
  const int qw0 = qt * 64 + wave * 16;

  __shared__ __bf16 kbuf[32][kHD];        // K rows   [t][d]   (B-frag of Q*K^T)
  __shared__ __bf16 vbuf[kHD][32];        // V^T      [d][t]   (B-frag of P*V)
  __shared__ __bf16 pbuf[4][16][32];      // per-wave P tile -> A-frag transpose

  Frag qf[4];
  const __bf16* qbase = Q + ((size_t)(b * kHQ + h) * kS + qw0) * kHD;
  #pragma unroll
  for (int dc = 0; dc < 4; ++dc) {
    qf[dc].q[0] = *(const uint4*)(qbase + (size_t)mloc * kHD + dc * 32 + kh * 8);
    qf[dc].q[1] = *(const uint4*)(qbase + (size_t)mloc * kHD + dc * 32 + 16 + kh * 8);
  }

  v8f zf = {};
  float mrun[8], lrun[8];
  v8f acc[8];
  int segq[8];
  #pragma unroll
  for (int r = 0; r < 8; ++r) {
    mrun[r] = -3.0e38f; lrun[r] = 0.f;
    segq[r] = seg[b * kS + qw0 + kh * 8 + r];
  }
  #pragma unroll
  for (int d = 0; d < 8; ++d) acc[d] = zf;

  const __bf16* kbase = Kc + (size_t)(b * kHKV + hk) * kS * kHD;
  const __bf16* vbase = Vc + (size_t)(b * kHKV + hk) * kS * kHD;
  const int tend = qt * 64 + 64;

  for (int t0 = 0; t0 < tend; t0 += 32) {
    __syncthreads();
    {   // stage K chunk row-major, V chunk transposed
      const int trow = threadIdx.x >> 2;
      const int d0 = (threadIdx.x & 3) * 32;
      const __bf16* kg = kbase + (size_t)(t0 + trow) * kHD + d0;
      const __bf16* vg = vbase + (size_t)(t0 + trow) * kHD + d0;
      #pragma unroll
      for (int u = 0; u < 4; ++u)
        *(uint4*)&kbuf[trow][d0 + u * 8] = *(const uint4*)(kg + u * 8);
      #pragma unroll
      for (int u = 0; u < 32; ++u) vbuf[d0 + u][trow] = vg[u];
    }
    __syncthreads();

    v8f s0 = zf, s1 = zf;
    #pragma unroll
    for (int dc = 0; dc < 4; ++dc) {
      Frag k0f, k1f;
      k0f.q[0] = *(const uint4*)&kbuf[mloc][dc * 32 + kh * 8];
      k0f.q[1] = *(const uint4*)&kbuf[mloc][dc * 32 + 16 + kh * 8];
      k1f.q[0] = *(const uint4*)&kbuf[16 + mloc][dc * 32 + kh * 8];
      k1f.q[1] = *(const uint4*)&kbuf[16 + mloc][dc * 32 + 16 + kh * 8];
      s0 = wmma_bf16(qf[dc].v, k0f.v, s0);
      s1 = wmma_bf16(qf[dc].v, k1f.v, s1);
    }

    const int tc0 = t0 + mloc, tc1 = tc0 + 16;
    const int st0 = seg[b * kS + tc0], st1 = seg[b * kS + tc1];
    float p0[8], p1[8];
    #pragma unroll
    for (int r = 0; r < 8; ++r) {
      const int qrow = qw0 + kh * 8 + r;
      const float v0 = (tc0 <= qrow && st0 == segq[r]) ? s0[r] : -1e9f;
      const float v1 = (tc1 <= qrow && st1 == segq[r]) ? s1[r] : -1e9f;
      float cm = fmaxf(v0, v1);
      cm = fmaxf(cm, __shfl_xor(cm, 1, 32));
      cm = fmaxf(cm, __shfl_xor(cm, 2, 32));
      cm = fmaxf(cm, __shfl_xor(cm, 4, 32));
      cm = fmaxf(cm, __shfl_xor(cm, 8, 32));
      const float nm = fmaxf(mrun[r], cm);
      const float al = __expf(mrun[r] - nm);
      p0[r] = __expf(v0 - nm);
      p1[r] = __expf(v1 - nm);
      float ps = p0[r] + p1[r];
      ps += __shfl_xor(ps, 1, 32);
      ps += __shfl_xor(ps, 2, 32);
      ps += __shfl_xor(ps, 4, 32);
      ps += __shfl_xor(ps, 8, 32);
      lrun[r] = lrun[r] * al + ps;
      mrun[r] = nm;
      #pragma unroll
      for (int d = 0; d < 8; ++d) acc[d][r] *= al;
    }
    #pragma unroll
    for (int r = 0; r < 8; ++r) {        // C layout -> row-major P tile in LDS
      pbuf[wave][kh * 8 + r][mloc]      = (__bf16)p0[r];
      pbuf[wave][kh * 8 + r][16 + mloc] = (__bf16)p1[r];
    }
    __syncthreads();
    Frag pf;                             // reload P as an A fragment (16x32 over t)
    pf.q[0] = *(const uint4*)&pbuf[wave][mloc][kh * 8];
    pf.q[1] = *(const uint4*)&pbuf[wave][mloc][16 + kh * 8];
    #pragma unroll
    for (int d = 0; d < 8; ++d) {
      Frag vf;
      vf.q[0] = *(const uint4*)&vbuf[d * 16 + mloc][kh * 8];
      vf.q[1] = *(const uint4*)&vbuf[d * 16 + mloc][16 + kh * 8];
      acc[d] = wmma_bf16(pf.v, vf.v, acc[d]);
    }
  }

  #pragma unroll
  for (int d = 0; d < 8; ++d)
    #pragma unroll
    for (int r = 0; r < 8; ++r) {
      const size_t orow = (size_t)b * kS + qw0 + kh * 8 + r;
      ctx[orow * (kHQ * kHD) + h * kHD + d * 16 + mloc] =
          (__bf16)(acc[d][r] / lrun[r]);
    }
}

// ---------------------------------------------------------------------------
// Elementwise helpers
// ---------------------------------------------------------------------------
__global__ __launch_bounds__(256)
void add_f32_kernel(const float* __restrict__ a, const float* __restrict__ b,
                    float* __restrict__ o) {
  const size_t i = (size_t)blockIdx.x * 256 + threadIdx.x;
  o[i] = a[i] + b[i];
}

__global__ __launch_bounds__(256)
void copy_f32_kernel(const float* __restrict__ a, float* __restrict__ o) {
  const size_t i = (size_t)blockIdx.x * 256 + threadIdx.x;
  o[i] = a[i];
}

__global__ __launch_bounds__(256)
void silu_mul_to_bf16_kernel(const float* __restrict__ a, const float* __restrict__ b,
                             __bf16* __restrict__ o) {
  const size_t i = (size_t)blockIdx.x * 256 + threadIdx.x;
  const float x = a[i];
  o[i] = (__bf16)((x / (1.f + __expf(-x))) * b[i]);
}

__global__ __launch_bounds__(256)
void moe_accum_kernel(float* __restrict__ out, const float* __restrict__ x,
                      const float* __restrict__ mw, int e) {
  const size_t i = (size_t)blockIdx.x * 256 + threadIdx.x;
  const int row = (int)(i / kD);
  out[i] += mw[row * kE + e] * x[i];
}

// ---------------------------------------------------------------------------
// Gate logits + top-2 softmax routing weights (one block per token)
// ---------------------------------------------------------------------------
__global__ __launch_bounds__(256)
void gate_topk_kernel(const __bf16* __restrict__ h, const float* __restrict__ gw,
                      float* __restrict__ mw) {
  const int row = blockIdx.x;
  float p[kE];
  #pragma unroll
  for (int e = 0; e < kE; ++e) p[e] = 0.f;
  for (int d = threadIdx.x; d < kD; d += 256) {
    const float hv = (float)h[(size_t)row * kD + d];
    #pragma unroll
    for (int e = 0; e < kE; ++e) p[e] += hv * gw[d * kE + e];
  }
  __shared__ float red[kE][256];
  #pragma unroll
  for (int e = 0; e < kE; ++e) red[e][threadIdx.x] = p[e];
  __syncthreads();
  for (int off = 128; off >= 1; off >>= 1) {
    if (threadIdx.x < off)
      #pragma unroll
      for (int e = 0; e < kE; ++e)
        red[e][threadIdx.x] += red[e][threadIdx.x + off];
    __syncthreads();
  }
  if (threadIdx.x == 0) {
    float v[kE];
    #pragma unroll
    for (int e = 0; e < kE; ++e) v[e] = red[e][0];
    int e1 = 0;
    for (int e = 1; e < kE; ++e) if (v[e] > v[e1]) e1 = e;       // first max
    int e2 = (e1 == 0) ? 1 : 0;
    for (int e = 0; e < kE; ++e) if (e != e1 && v[e] > v[e2]) e2 = e;
    const float w2 = __expf(v[e2] - v[e1]);
    const float z = 1.f + w2;
    for (int e = 0; e < kE; ++e)
      mw[row * kE + e] = (e == e1) ? (1.f / z) : (e == e2) ? (w2 / z) : 0.f;
  }
}

// ---------------------------------------------------------------------------
// Launch: inputs in setup_inputs() order; all work on `stream`.
// Workspace budget ~= 480 MB (bf16 weights/activations + fp32 temporaries).
// ---------------------------------------------------------------------------
extern "C" void kernel_launch(void* const* d_in, const int* in_sizes, int n_in,
                              void* d_out, int out_size, void* d_ws, size_t ws_size,
                              hipStream_t stream) {
  (void)in_sizes; (void)n_in; (void)out_size; (void)ws_size;
  const float* x       = (const float*)d_in[0];
  const float* preln   = (const float*)d_in[1];
  const float* postln  = (const float*)d_in[2];
  const float* wq      = (const float*)d_in[3];
  const float* wk      = (const float*)d_in[4];
  const float* wv      = (const float*)d_in[5];
  const float* wo_attn = (const float*)d_in[6];
  const float* gate_w  = (const float*)d_in[7];
  const float* wi0     = (const float*)d_in[8];
  const float* wi1     = (const float*)d_in[9];
  const float* wo_mlp  = (const float*)d_in[10];
  const int*   segids  = (const int*)d_in[11];
  const int*   pos     = (const int*)d_in[12];
  float* out = (float*)d_out;

  char* W = (char*)d_ws;
  size_t off = 0;
  auto take = [&](size_t bytes) {
    char* p = W + off;
    off = (off + bytes + 255) & ~(size_t)255;
    return p;
  };
  __bf16* lnx   = (__bf16*)take((size_t)kT * kD * 2);
  __bf16* wqkvT = (__bf16*)take((size_t)kQKV * kD * 2);
  float*  qkv   = (float*) take((size_t)kT * kQKV * 4);
  __bf16* qb    = (__bf16*)take((size_t)kT * kHQ * kHD * 2);
  __bf16* kb    = (__bf16*)take((size_t)kT * kHKV * kHD * 2);
  __bf16* vb    = (__bf16*)take((size_t)kT * kHKV * kHD * 2);
  __bf16* ctx   = (__bf16*)take((size_t)kT * kD * 2);
  __bf16* woT   = (__bf16*)take((size_t)kD * kD * 2);
  float*  attno = (float*) take((size_t)kT * kD * 4);
  float*  inter = (float*) take((size_t)kT * kD * 4);
  __bf16* hb    = (__bf16*)take((size_t)kT * kD * 2);
  float*  mw    = (float*) take((size_t)kT * kE * 4);
  __bf16* wi0T  = (__bf16*)take((size_t)kF * kD * 2);
  __bf16* wi1T  = (__bf16*)take((size_t)kF * kD * 2);
  __bf16* womT  = (__bf16*)take((size_t)kD * kF * 2);
  float*  t0    = (float*) take((size_t)kT * kF * 4);
  float*  t1    = (float*) take((size_t)kT * kF * 4);
  __bf16* act   = (__bf16*)take((size_t)kT * kF * 2);
  float*  t2    = (float*) take((size_t)kT * kD * 4);

  const dim3 tb(32, 8);

  // --- pre-attention RMSNorm ---
  rmsnorm_to_bf16_kernel<<<kT, 256, 0, stream>>>(x, preln, lnx, kD);

  // --- fused QKV weights: W^T bf16, rows [0,2048)=Q, [2048,3072)=K, [3072,4096)=V
  transpose_f32_to_bf16_kernel<<<dim3(kD / 32, kD / 32), tb, 0, stream>>>(
      wq, wqkvT, kD, kD);
  transpose_f32_to_bf16_kernel<<<dim3((kHKV * kHD) / 32, kD / 32), tb, 0, stream>>>(
      wk, wqkvT + (size_t)kD * kD, kD, kHKV * kHD);
  transpose_f32_to_bf16_kernel<<<dim3((kHKV * kHD) / 32, kD / 32), tb, 0, stream>>>(
      wv, wqkvT + (size_t)(kD + kHKV * kHD) * kD, kD, kHKV * kHD);

  // --- QKV projection ---
  gemm_bf16_wmma_kernel<<<dim3(kQKV / 128, kT / 128), 256, 0, stream>>>(
      lnx, wqkvT, qkv, kT, kQKV, kD);

  // --- RoPE + q scaling + head split ---
  rope_split_kernel<<<kT, 256, 0, stream>>>(qkv, pos, qb, kb, vb);

  // --- flash attention ---
  flash_attn_kernel<<<dim3(kB * kHQ, kS / 64), 128, 0, stream>>>(qb, kb, vb, segids, ctx);

  // --- output projection + residual ---
  transpose_f32_to_bf16_kernel<<<dim3(kD / 32, kD / 32), tb, 0, stream>>>(
      wo_attn, woT, kD, kD);
  gemm_bf16_wmma_kernel<<<dim3(kD / 128, kT / 128), 256, 0, stream>>>(
      ctx, woT, attno, kT, kD, kD);
  add_f32_kernel<<<(size_t)kT * kD / 256, 256, 0, stream>>>(x, attno, inter);

  // --- post-attention RMSNorm + routing ---
  rmsnorm_to_bf16_kernel<<<kT, 256, 0, stream>>>(inter, postln, hb, kD);
  gate_topk_kernel<<<kT, 256, 0, stream>>>(hb, gate_w, mw);

  // --- out = intermediate, then accumulate experts ---
  copy_f32_kernel<<<(size_t)kT * kD / 256, 256, 0, stream>>>(inter, out);

  for (int e = 0; e < kE; ++e) {
    transpose_f32_to_bf16_kernel<<<dim3(kF / 32, kD / 32), tb, 0, stream>>>(
        wi0 + (size_t)e * kD * kF, wi0T, kD, kF);
    transpose_f32_to_bf16_kernel<<<dim3(kF / 32, kD / 32), tb, 0, stream>>>(
        wi1 + (size_t)e * kD * kF, wi1T, kD, kF);
    transpose_f32_to_bf16_kernel<<<dim3(kD / 32, kF / 32), tb, 0, stream>>>(
        wo_mlp + (size_t)e * kF * kD, womT, kF, kD);
    gemm_bf16_wmma_kernel<<<dim3(kF / 128, kT / 128), 256, 0, stream>>>(
        hb, wi0T, t0, kT, kF, kD);
    gemm_bf16_wmma_kernel<<<dim3(kF / 128, kT / 128), 256, 0, stream>>>(
        hb, wi1T, t1, kT, kF, kD);
    silu_mul_to_bf16_kernel<<<(size_t)kT * kF / 256, 256, 0, stream>>>(t0, t1, act);
    gemm_bf16_wmma_kernel<<<dim3(kD / 128, kT / 128), 256, 0, stream>>>(
        act, womT, t2, kT, kD, kF);
    moe_accum_kernel<<<(size_t)kT * kD / 256, 256, 0, stream>>>(out, t2, mw, e);
  }
}
